// SemanticsModulatedAttention_8057358647539
// MI455X (gfx1250) — matrix operand order repaired and checked by reading, not compiled, not run on hardware
//
#include <hip/hip_runtime.h>

// ---------------------------------------------------------------------------
// SemanticsModulatedAttention for MI455X (gfx1250, wave32, WMMA bf16)
// All matmuls routed through a batched WMMA GEMM (v_wmma_f32_16x16x32_bf16,
// f32 accumulate). Contiguous LDS staging uses gfx1250 async copies
// (global_load_async_to_lds_b128, ASYNCcnt). Operands bf16, accums f32.
// ---------------------------------------------------------------------------

#define B_  64
#define T_  512
#define D_  512
#define H_  8
#define HD_ 64
#define CT_ 77
#define CS_ 64
#define LD_ 64
#define TE_ 2048
#define NEGV_ (-1000000.0f)

typedef __attribute__((ext_vector_type(16))) __bf16 v16bf;
typedef __attribute__((ext_vector_type(8)))  float  v8f;
typedef unsigned short ushort_t;

__device__ __forceinline__ ushort_t f2bf(float f) {
    union { float f; unsigned u; } v; v.f = f;
    unsigned r = v.u + 0x7FFFu + ((v.u >> 16) & 1u);   // round-to-nearest-even
    return (ushort_t)(r >> 16);
}
__device__ __forceinline__ __bf16 us2bf(ushort_t s) {
    union { ushort_t s; __bf16 b; } o; o.s = s; return o.b;
}
__device__ __forceinline__ float sigm(float x) { return 1.0f / (1.0f + __expf(-x)); }

#define LDSLD 72   // LDS row stride (elements): 144B rows -> b128-aligned staging

// async 16B global -> LDS copy (gfx1250, tracked by ASYNCcnt)
__device__ __forceinline__ void async_g2l_b128(ushort_t* lds, const ushort_t* gsrc) {
    unsigned           ldsoff = (unsigned)(unsigned long long)lds;       // flat[31:0] == LDS offset
    unsigned long long ga     = (unsigned long long)gsrc;
    asm volatile("global_load_async_to_lds_b128 %0, %1, off"
                 :: "v"(ldsoff), "v"(ga) : "memory");
}
__device__ __forceinline__ void wait_async0() {
    asm volatile("s_wait_asynccnt 0x0" ::: "memory");
}

// A fragment (16x32 bf16, CDNA5 wave32 layout), k window starts at kbase
__device__ __forceinline__ v16bf fragA(const ushort_t* As, int ar, int kbase, int laneHi) {
    v16bf a;
    #pragma unroll
    for (int i = 0; i < 8; ++i) {
        int kk = kbase + (i < 4 ? 2 * i : 16 + 2 * (i - 4)) + laneHi * 8;
        union { unsigned u; ushort_t s[2]; } p;
        p.u = *(const unsigned*)&As[ar * LDSLD + kk];
        a[2 * i]     = us2bf(p.s[0]);
        a[2 * i + 1] = us2bf(p.s[1]);
    }
    return a;
}
// B fragment (32x16 bf16) from LDS stored [col][k]
__device__ __forceinline__ v16bf fragB(const ushort_t* Bs, int col, int kbase, int laneHi) {
    v16bf b;
    #pragma unroll
    for (int j = 0; j < 8; ++j) {
        int kk = kbase + (j < 4 ? 2 * j : 16 + 2 * (j - 4)) + laneHi * 8;
        union { unsigned u; ushort_t s[2]; } p;
        p.u = *(const unsigned*)&Bs[col * LDSLD + kk];
        b[2 * j]     = us2bf(p.s[0]);
        b[2 * j + 1] = us2bf(p.s[1]);
    }
    return b;
}

// ---------------------------------------------------------------------------
// Batched WMMA GEMM: C[z] = A[z](MxK) * B[z](KxN) (+bias) (*rowmask) [+=]
// A,B bf16; C f32 or bf16. z decomposed as (z/HH, z%HH) for strides.
// TA: A[m,k]=Asrc[k*lda+m]; TB: B[k,n]=Bsrc[n*ldb+k].
// Block: 256 threads (8 waves), 64x64 tile, BK=64 (4 WMMA per barrier).
// ---------------------------------------------------------------------------
template<int TA, int TB>
__global__ __launch_bounds__(256)
void wmma_gemm_bf16(const ushort_t* __restrict__ A, const ushort_t* __restrict__ Bm,
                    const float* __restrict__ bias,
                    float* __restrict__ Cf, ushort_t* __restrict__ Cb,
                    const float* __restrict__ rmask, long long sMb,
                    int M, int N, int K, int lda, int ldb, int ldc,
                    long long sAb, long long sAh, long long sBb, long long sBh,
                    long long sCb, long long sCh, int HH, int accum)
{
    __shared__ ushort_t As[64 * LDSLD];   // [row][k]
    __shared__ ushort_t Bs[64 * LDSLD];   // [col][k]

    const int tid  = threadIdx.x;
    const int z    = blockIdx.z;
    const int zb   = z / HH, zh = z % HH;
    const long long offA = (long long)zb * sAb + (long long)zh * sAh;
    const long long offB = (long long)zb * sBb + (long long)zh * sBh;
    const long long offC = (long long)zb * sCb + (long long)zh * sCh;

    const int m0 = blockIdx.y * 64;
    const int n0 = blockIdx.x * 64;

    const int lane   = tid & 31;
    const int l15    = lane & 15;
    const int laneHi = (lane >> 4) & 1;
    const int wv     = tid >> 5;       // wave id 0..7
    const int mt     = wv & 3;         // M subtile 0..3
    const int nt0    = (wv >> 2) * 32; // N base col: 0 or 32 (two tiles each)

    v8f acc0 = {}, acc1 = {};

    for (int k0 = 0; k0 < K; k0 += 64) {
        const bool full = (m0 + 64 <= M) && (n0 + 64 <= N) && (k0 + 64 <= K);
        if (full) {
            // ---------------- vectorized 16B staging ----------------
            #pragma unroll
            for (int it = 0; it < 2; ++it) {
                int v = it * 256 + tid;            // 0..511 vectors of 8 bf16
                if (TA == 0) {                      // contiguous: async copy to LDS
                    int r = v >> 3, kc = (v & 7) * 8;
                    async_g2l_b128(&As[r * LDSLD + kc],
                                   A + offA + (long long)(m0 + r) * lda + (k0 + kc));
                } else {                            // transpose via registers
                    int mc = (v & 7) * 8, kk = v >> 3;
                    const ushort_t* src = A + offA + (long long)(k0 + kk) * lda + (m0 + mc);
                    union { uint4 q; ushort_t s[8]; } u; u.q = *(const uint4*)src;
                    #pragma unroll
                    for (int j = 0; j < 8; ++j) As[(mc + j) * LDSLD + kk] = u.s[j];
                }
            }
            #pragma unroll
            for (int it = 0; it < 2; ++it) {
                int v = it * 256 + tid;
                if (TB == 1) {                      // source [N,K]: contiguous -> async
                    int c = v >> 3, kc = (v & 7) * 8;
                    async_g2l_b128(&Bs[c * LDSLD + kc],
                                   Bm + offB + (long long)(n0 + c) * ldb + (k0 + kc));
                } else {                            // source [K,N]: transpose via registers
                    int nc = (v & 7) * 8, kk = v >> 3;
                    const ushort_t* src = Bm + offB + (long long)(k0 + kk) * ldb + (n0 + nc);
                    union { uint4 q; ushort_t s[8]; } u; u.q = *(const uint4*)src;
                    #pragma unroll
                    for (int j = 0; j < 8; ++j) Bs[(nc + j) * LDSLD + kk] = u.s[j];
                }
            }
            if (TA == 0 || TB == 1) wait_async0();  // ASYNCcnt not covered by barrier
        } else {
            // ------- branchless guarded staging (edge tiles): clamp + select 0 -------
            for (int i = tid; i < 64 * 64; i += 256) {
                int r = i >> 6, kk = i & 63;
                int gm = m0 + r, gk = k0 + kk;
                int cm = gm < M ? gm : M - 1;
                int ck = gk < K ? gk : K - 1;
                ushort_t vv = A[offA + (TA ? (long long)ck * lda + cm
                                           : (long long)cm * lda + ck)];
                As[r * LDSLD + kk] = (gm < M && gk < K) ? vv : (ushort_t)0;
            }
            for (int i = tid; i < 64 * 64; i += 256) {
                int c = i >> 6, kk = i & 63;
                int gn = n0 + c, gk = k0 + kk;
                int cn = gn < N ? gn : N - 1;
                int ck = gk < K ? gk : K - 1;
                ushort_t vv = Bm[offB + (TB ? (long long)cn * ldb + ck
                                            : (long long)ck * ldb + cn)];
                Bs[c * LDSLD + kk] = (gn < N && gk < K) ? vv : (ushort_t)0;
            }
        }
        __syncthreads();

        if (k0 + 64 < K) {  // gfx1250 global_prefetch_b8 for next K tile
            __builtin_prefetch(&A[offA + (TA ? (long long)(k0 + 64) * lda + m0
                                             : (long long)m0 * lda + (k0 + 64))], 0, 1);
            __builtin_prefetch(&Bm[offB + (TB ? (long long)n0 * ldb + (k0 + 64)
                                              : (long long)(k0 + 64) * ldb + n0)], 0, 1);
        }

        const int ar  = mt * 16 + l15;
        const int bc0 = nt0 + l15;
        v16bf a0  = fragA(As, ar, 0,  laneHi);
        v16bf a1  = fragA(As, ar, 32, laneHi);
        v16bf b00 = fragB(Bs, bc0,      0,  laneHi);
        v16bf b01 = fragB(Bs, bc0,      32, laneHi);
        v16bf b10 = fragB(Bs, bc0 + 16, 0,  laneHi);
        v16bf b11 = fragB(Bs, bc0 + 16, 32, laneHi);

        acc0 = __builtin_amdgcn_wmma_f32_16x16x32_bf16(false, a0, false, b00,
                                                       (short)0, acc0, false, false);
        acc0 = __builtin_amdgcn_wmma_f32_16x16x32_bf16(false, a1, false, b01,
                                                       (short)0, acc0, false, false);
        acc1 = __builtin_amdgcn_wmma_f32_16x16x32_bf16(false, a0, false, b10,
                                                       (short)0, acc1, false, false);
        acc1 = __builtin_amdgcn_wmma_f32_16x16x32_bf16(false, a1, false, b11,
                                                       (short)0, acc1, false, false);
        __syncthreads();
    }

    // ---- epilogue (C/D layout: VGPR r -> row mt*16 + laneHi*8 + r, col = l15) ----
    #pragma unroll
    for (int r = 0; r < 8; ++r) {
        int row = m0 + mt * 16 + laneHi * 8 + r;
        if (row >= M) continue;
        float rm = rmask ? rmask[(long long)zb * sMb + row] : 1.0f;
        int col0 = n0 + nt0 + l15;
        int col1 = col0 + 16;
        if (col0 < N) {
            float v = acc0[r] + (bias ? bias[col0] : 0.0f);
            v *= rm;
            long long ci = offC + (long long)row * ldc + col0;
            if (Cf) { if (accum) Cf[ci] += v; else Cf[ci] = v; }
            else    { Cb[ci] = f2bf(v); }
        }
        if (col1 < N) {
            float v = acc1[r] + (bias ? bias[col1] : 0.0f);
            v *= rm;
            long long ci = offC + (long long)row * ldc + col1;
            if (Cf) { if (accum) Cf[ci] += v; else Cf[ci] = v; }
            else    { Cb[ci] = f2bf(v); }
        }
    }
}

// ---------------------------------------------------------------------------
// LayerNorm over last dim -> bf16. One block per row.
// ---------------------------------------------------------------------------
__global__ void ln_bf16_kernel(const float* __restrict__ x, const float* __restrict__ g,
                               const float* __restrict__ b, ushort_t* __restrict__ out, int dim)
{
    long long row = blockIdx.x;
    const float* xr = x + row * dim;
    __shared__ float r1[256], r2[256];
    float s = 0.f, s2 = 0.f;
    for (int i = threadIdx.x; i < dim; i += blockDim.x) { float v = xr[i]; s += v; s2 += v * v; }
    r1[threadIdx.x] = s; r2[threadIdx.x] = s2; __syncthreads();
    for (int st = (int)blockDim.x >> 1; st > 0; st >>= 1) {
        if ((int)threadIdx.x < st) { r1[threadIdx.x] += r1[threadIdx.x + st];
                                     r2[threadIdx.x] += r2[threadIdx.x + st]; }
        __syncthreads();
    }
    float mean = r1[0] / dim;
    float var  = r2[0] / dim - mean * mean;
    float inv  = rsqrtf(var + 1e-5f);
    for (int i = threadIdx.x; i < dim; i += blockDim.x)
        out[row * dim + i] = f2bf((xr[i] - mean) * inv * g[i] + b[i]);
}

// ---------------------------------------------------------------------------
// Row softmax (axis=-1) -> bf16. Optional additive concat mask:
//   n <  Cc : (1-cmask[b*Cm+n])*NEG      (cond part)
//   n >= Cc : (1-smask[b*Tm+(n-Cc)])*NEG (x part)
// ---------------------------------------------------------------------------
__global__ void row_softmax_bf16(const float* __restrict__ in, ushort_t* __restrict__ out,
                                 int dim, int maskMode, int Cc, int rowsPerB, int batchBase,
                                 int Tm, int Cm,
                                 const float* __restrict__ cmask, const float* __restrict__ smask)
{
    long long row = blockIdx.x;
    const float* xr = in + row * dim;
    int b = batchBase + (int)(row / rowsPerB);
    __shared__ float red[256];
    auto mval = [&](int n) -> float {
        if (!maskMode) return 0.0f;
        return (n < Cc) ? (1.0f - cmask[(long long)b * Cm + n]) * NEGV_
                        : (1.0f - smask[(long long)b * Tm + (n - Cc)]) * NEGV_;
    };
    float m = -3.4e38f;
    for (int i = threadIdx.x; i < dim; i += blockDim.x) m = fmaxf(m, xr[i] + mval(i));
    red[threadIdx.x] = m; __syncthreads();
    for (int st = (int)blockDim.x >> 1; st > 0; st >>= 1) {
        if ((int)threadIdx.x < st) red[threadIdx.x] = fmaxf(red[threadIdx.x], red[threadIdx.x + st]);
        __syncthreads();
    }
    m = red[0]; __syncthreads();
    float s = 0.f;
    for (int i = threadIdx.x; i < dim; i += blockDim.x) s += __expf(xr[i] + mval(i) - m);
    red[threadIdx.x] = s; __syncthreads();
    for (int st = (int)blockDim.x >> 1; st > 0; st >>= 1) {
        if ((int)threadIdx.x < st) red[threadIdx.x] += red[threadIdx.x + st];
        __syncthreads();
    }
    float inv = 1.0f / red[0];
    for (int i = threadIdx.x; i < dim; i += blockDim.x)
        out[row * dim + i] = f2bf(__expf(xr[i] + mval(i) - m) * inv);
}

// ---------------------------------------------------------------------------
// Column softmax over sequence axis (axis=1) of [B'][ntot][D] -> bf16.
// Rows n>=c get additive (1-src_mask)*NEG.
// ---------------------------------------------------------------------------
__global__ void col_softmax_bf16(const float* __restrict__ in, ushort_t* __restrict__ out,
                                 int ntot, int c, int D, const float* __restrict__ smask,
                                 int Tm, int batchBase)
{
    int d = blockIdx.x * blockDim.x + threadIdx.x;
    int b = blockIdx.y;
    if (d >= D) return;
    const float* base = in + (long long)b * ntot * D;
    const float* mk   = smask + (long long)(b + batchBase) * Tm;
    float m = -3.4e38f;
    for (int n = 0; n < ntot; ++n) {
        float v = base[(long long)n * D + d];
        if (n >= c) v += (1.0f - mk[n - c]) * NEGV_;
        m = fmaxf(m, v);
    }
    float s = 0.f;
    for (int n = 0; n < ntot; ++n) {
        float v = base[(long long)n * D + d];
        if (n >= c) v += (1.0f - mk[n - c]) * NEGV_;
        s += __expf(v - m);
    }
    float inv = 1.0f / s;
    ushort_t* ob = out + (long long)b * ntot * D;
    for (int n = 0; n < ntot; ++n) {
        float v = base[(long long)n * D + d];
        if (n >= c) v += (1.0f - mk[n - c]) * NEGV_;
        ob[(long long)n * D + d] = f2bf(__expf(v - m) * inv);
    }
}

// ---------------------------------------------------------------------------
// Elementwise kernels
// ---------------------------------------------------------------------------
__global__ void cvt_bf16(const float* __restrict__ in, ushort_t* __restrict__ out, long long n)
{
    long long i = (long long)blockIdx.x * blockDim.x + threadIdx.x;
    if (i < n) out[i] = f2bf(in[i]);
}
__global__ void silu_bf16(const float* __restrict__ in, ushort_t* __restrict__ out, long long n)
{
    long long i = (long long)blockIdx.x * blockDim.x + threadIdx.x;
    if (i < n) { float v = in[i]; out[i] = f2bf(v * sigm(v)); }
}
// stylization: h = ln(q_)*(1+scale)+shift ; out = bf16(silu(h))
__global__ void modulate_silu_bf16(const float* __restrict__ q_, const float* __restrict__ e,
                                   const float* __restrict__ ng, const float* __restrict__ nb,
                                   ushort_t* __restrict__ out, int T)
{
    long long row = blockIdx.x;           // b*T + t
    int b = (int)(row / T);
    const float* xr = q_ + row * D_;
    __shared__ float r1[256], r2[256];
    float s = 0.f, s2 = 0.f;
    for (int i = threadIdx.x; i < D_; i += blockDim.x) { float v = xr[i]; s += v; s2 += v * v; }
    r1[threadIdx.x] = s; r2[threadIdx.x] = s2; __syncthreads();
    for (int st = (int)blockDim.x >> 1; st > 0; st >>= 1) {
        if ((int)threadIdx.x < st) { r1[threadIdx.x] += r1[threadIdx.x + st];
                                     r2[threadIdx.x] += r2[threadIdx.x + st]; }
        __syncthreads();
    }
    float mean = r1[0] / D_;
    float var  = r2[0] / D_ - mean * mean;
    float inv  = rsqrtf(var + 1e-5f);
    const float* eb = e + (long long)b * (2 * D_);
    for (int i = threadIdx.x; i < D_; i += blockDim.x) {
        float nx = (xr[i] - mean) * inv * ng[i] + nb[i];
        float h  = nx * (1.0f + eb[i]) + eb[D_ + i];
        out[row * D_ + i] = f2bf(h * sigm(h));
    }
}
__global__ void add_kernel(const float* __restrict__ a, const float* __restrict__ b,
                           float* __restrict__ out, long long n)
{
    long long i = (long long)blockIdx.x * blockDim.x + threadIdx.x;
    if (i < n) out[i] = a[i] + b[i];
}

// ---------------------------------------------------------------------------
// Host side
// ---------------------------------------------------------------------------
static inline int ceilDiv(int a, int b) { return (a + b - 1) / b; }

static void launch_gemm(hipStream_t st, const ushort_t* A, const ushort_t* Bm, const float* bias,
                        float* Cf, ushort_t* Cb, const float* rmask, long long sMb,
                        int M, int N, int K, int lda, int ldb, int ldc,
                        long long sAb, long long sAh, long long sBb, long long sBh,
                        long long sCb, long long sCh, int HH, int nbatch,
                        int tA, int tB, int acc)
{
    dim3 g(ceilDiv(N, 64), ceilDiv(M, 64), nbatch);
    if (tA)
        wmma_gemm_bf16<1, 0><<<g, 256, 0, st>>>(A, Bm, bias, Cf, Cb, rmask, sMb,
            M, N, K, lda, ldb, ldc, sAb, sAh, sBb, sBh, sCb, sCh, HH, acc);
    else if (tB)
        wmma_gemm_bf16<0, 1><<<g, 256, 0, st>>>(A, Bm, bias, Cf, Cb, rmask, sMb,
            M, N, K, lda, ldb, ldc, sAb, sAh, sBb, sBh, sCb, sCh, HH, acc);
    else
        wmma_gemm_bf16<0, 0><<<g, 256, 0, st>>>(A, Bm, bias, Cf, Cb, rmask, sMb,
            M, N, K, lda, ldb, ldc, sAb, sAh, sBb, sBh, sCb, sCh, HH, acc);
}

extern "C" void kernel_launch(void* const* d_in, const int* in_sizes, int n_in,
                              void* d_out, int out_size, void* d_ws, size_t ws_size,
                              hipStream_t stream)
{
    (void)in_sizes; (void)n_in; (void)out_size; (void)ws_size;
    // ---- input pointers (jax pytree flatten order: top-level insertion order,
    //      nested param dicts alphabetical by key) ----
    const float* x         = (const float*)d_in[0];
    const float* text_emb  = (const float*)d_in[1];
    const float* stick_emb = (const float*)d_in[2];
    const float* other_emb = (const float*)d_in[3];
    const float* src_mask  = (const float*)d_in[4];
    const float* stick_mask= (const float*)d_in[5];
    const float* locus_emb = (const float*)d_in[6];
    const float* p_norm_b  = (const float*)d_in[7];
    const float* p_norm_g  = (const float*)d_in[8];
    const float* p_e_W     = (const float*)d_in[9];
    const float* p_e_b     = (const float*)d_in[10];
    const float* p_n_b     = (const float*)d_in[11];
    const float* p_n_g     = (const float*)d_in[12];
    const float* p_o_W     = (const float*)d_in[13];
    const float* p_o_b     = (const float*)d_in[14];
    const float* p_q_W     = (const float*)d_in[15];
    const float* p_q_b     = (const float*)d_in[16];
    const float* p_ql_W    = (const float*)d_in[17];
    const float* p_ql_b    = (const float*)d_in[18];
    // stick attn params (alphabetical)
    const float* s_kc_W = (const float*)d_in[19]; const float* s_kc_b = (const float*)d_in[20];
    const float* s_kx_W = (const float*)d_in[21]; const float* s_kx_b = (const float*)d_in[22];
    const float* s_nc_b = (const float*)d_in[23]; const float* s_nc_g = (const float*)d_in[24];
    const float* s_nx_b = (const float*)d_in[25]; const float* s_nx_g = (const float*)d_in[26];
    const float* s_py_W = (const float*)d_in[27]; const float* s_py_b = (const float*)d_in[28];
    const float* s_vc_W = (const float*)d_in[29]; const float* s_vc_b = (const float*)d_in[30];
    const float* s_vx_W = (const float*)d_in[31]; const float* s_vx_b = (const float*)d_in[32];
    // text attn params
    const float* t_kc_W = (const float*)d_in[33]; const float* t_kc_b = (const float*)d_in[34];
    const float* t_kx_W = (const float*)d_in[35]; const float* t_kx_b = (const float*)d_in[36];
    const float* t_nc_b = (const float*)d_in[37]; const float* t_nc_g = (const float*)d_in[38];
    const float* t_nx_b = (const float*)d_in[39]; const float* t_nx_g = (const float*)d_in[40];
    const float* t_py_W = (const float*)d_in[41]; const float* t_py_b = (const float*)d_in[42];
    const float* t_vc_W = (const float*)d_in[43]; const float* t_vc_b = (const float*)d_in[44];
    const float* t_vx_W = (const float*)d_in[45]; const float* t_vx_b = (const float*)d_in[46];
    // y (self) attn params (no cond)
    const float* y_kx_W = (const float*)d_in[47]; const float* y_kx_b = (const float*)d_in[48];
    const float* y_nx_b = (const float*)d_in[49]; const float* y_nx_g = (const float*)d_in[50];
    const float* y_py_W = (const float*)d_in[51]; const float* y_py_b = (const float*)d_in[52];
    const float* y_vx_W = (const float*)d_in[53]; const float* y_vx_b = (const float*)d_in[54];

    const int NT = T_ + CT_;   // 589
    const int NS = T_ + CS_;   // 576

    // ---- workspace arena ----
    char* wsp = (char*)d_ws;
    size_t cur = 0;
    auto alloc = [&](size_t bytes) -> void* {
        void* p = wsp + cur;
        cur += (bytes + 255) & ~(size_t)255;
        return p;
    };
    // persistent: bf16 weights, query f32, bf16 copy of query[16:48]
    ushort_t* wb_q    = (ushort_t*)alloc((size_t)D_ * D_ * 2);
    ushort_t* wb_ql   = (ushort_t*)alloc((size_t)LD_ * D_ * 2);
    ushort_t* wb_t_kc = (ushort_t*)alloc((size_t)256 * D_ * 2);
    ushort_t* wb_t_kx = (ushort_t*)alloc((size_t)D_ * D_ * 2);
    ushort_t* wb_t_vc = (ushort_t*)alloc((size_t)256 * D_ * 2);
    ushort_t* wb_t_vx = (ushort_t*)alloc((size_t)D_ * D_ * 2);
    ushort_t* wb_t_py = (ushort_t*)alloc((size_t)D_ * D_ * 2);
    ushort_t* wb_s_kc = (ushort_t*)alloc((size_t)256 * D_ * 2);
    ushort_t* wb_s_kx = (ushort_t*)alloc((size_t)D_ * D_ * 2);
    ushort_t* wb_s_vc = (ushort_t*)alloc((size_t)256 * D_ * 2);
    ushort_t* wb_s_vx = (ushort_t*)alloc((size_t)D_ * D_ * 2);
    ushort_t* wb_s_py = (ushort_t*)alloc((size_t)D_ * D_ * 2);
    ushort_t* wb_y_kx = (ushort_t*)alloc((size_t)D_ * D_ * 2);
    ushort_t* wb_y_vx = (ushort_t*)alloc((size_t)D_ * D_ * 2);
    ushort_t* wb_y_py = (ushort_t*)alloc((size_t)D_ * D_ * 2);
    ushort_t* wb_e    = (ushort_t*)alloc((size_t)TE_ * 2 * D_ * 2);
    ushort_t* wb_o    = (ushort_t*)alloc((size_t)D_ * D_ * 2);
    float*    qf      = (float*)   alloc((size_t)B_ * T_ * D_ * 4);
    ushort_t* s_qb    = (ushort_t*)alloc((size_t)32 * T_ * D_ * 2);
    size_t phaseBase = cur;

    auto cvt = [&](const float* in, ushort_t* out, long long n) {
        cvt_bf16<<<dim3((unsigned)((n + 255) / 256)), 256, 0, stream>>>(in, out, n);
    };

    // ---- weight conversions to bf16 ----
    cvt(p_q_W,  wb_q,    (long long)D_ * D_);
    cvt(p_ql_W, wb_ql,   (long long)LD_ * D_);
    cvt(t_kc_W, wb_t_kc, (long long)256 * D_);
    cvt(t_kx_W, wb_t_kx, (long long)D_ * D_);
    cvt(t_vc_W, wb_t_vc, (long long)256 * D_);
    cvt(t_vx_W, wb_t_vx, (long long)D_ * D_);
    cvt(t_py_W, wb_t_py, (long long)D_ * D_);
    cvt(s_kc_W, wb_s_kc, (long long)256 * D_);
    cvt(s_kx_W, wb_s_kx, (long long)D_ * D_);
    cvt(s_vc_W, wb_s_vc, (long long)256 * D_);
    cvt(s_vx_W, wb_s_vx, (long long)D_ * D_);
    cvt(s_py_W, wb_s_py, (long long)D_ * D_);
    cvt(y_kx_W, wb_y_kx, (long long)D_ * D_);
    cvt(y_vx_W, wb_y_vx, (long long)D_ * D_);
    cvt(y_py_W, wb_y_py, (long long)D_ * D_);
    cvt(p_e_W,  wb_e,    (long long)TE_ * 2 * D_);
    cvt(p_o_W,  wb_o,    (long long)D_ * D_);

    // ================= phase group 1: query + text + stick =================
    ushort_t* a_nx   = (ushort_t*)alloc((size_t)B_ * T_ * D_ * 2);
    ushort_t* tp_nx  = (ushort_t*)alloc((size_t)32 * T_ * D_ * 2);
    ushort_t* tp_nc  = (ushort_t*)alloc((size_t)32 * CT_ * 256 * 2);
    float*    tp_key = (float*)   alloc((size_t)32 * NT * D_ * 4);
    ushort_t* tp_v   = (ushort_t*)alloc((size_t)32 * NT * D_ * 2);
    ushort_t* tp_ks  = (ushort_t*)alloc((size_t)32 * NT * D_ * 2);
    ushort_t* tp_qs  = (ushort_t*)alloc((size_t)32 * T_ * D_ * 2);
    ushort_t* tp_at  = (ushort_t*)alloc((size_t)256 * HD_ * HD_ * 2);
    ushort_t* tp_y   = (ushort_t*)alloc((size_t)32 * T_ * D_ * 2);
    float*    sp_x   = (float*)   alloc((size_t)32 * T_ * D_ * 4);
    ushort_t* sp_loc = (ushort_t*)alloc((size_t)32 * T_ * LD_ * 2);
    ushort_t* sp_nx  = (ushort_t*)alloc((size_t)32 * T_ * D_ * 2);
    ushort_t* sp_nc  = (ushort_t*)alloc((size_t)32 * CS_ * 256 * 2);
    ushort_t* sp_k   = (ushort_t*)alloc((size_t)32 * NS * D_ * 2);
    ushort_t* sp_v   = (ushort_t*)alloc((size_t)32 * NS * D_ * 2);
    float*    sp_sc  = (float*)   alloc((size_t)16 * T_ * NS * 4);  // z-chunk
    ushort_t* sp_a   = (ushort_t*)alloc((size_t)16 * T_ * NS * 2);
    ushort_t* sp_y   = (ushort_t*)alloc((size_t)32 * T_ * D_ * 2);

    // query = ln(x, norm) @ q_W + q_b
    ln_bf16_kernel<<<B_ * T_, 256, 0, stream>>>(x, p_norm_g, p_norm_b, a_nx, D_);
    launch_gemm(stream, a_nx, wb_q, p_q_b, qf, nullptr, nullptr, 0,
                B_ * T_, D_, D_, D_, D_, D_, 0, 0, 0, 0, 0, 0, 1, 1, 0, 0, 0);
    // bf16 snapshot of original query[16:48] for std attention
    cvt(qf + (long long)16 * T_ * D_, s_qb, (long long)32 * T_ * D_);

    // ---------------- text branch (linear attention, b=0..31) ----------------
    ln_bf16_kernel<<<32 * T_, 256, 0, stream>>>(x, t_nx_g, t_nx_b, tp_nx, D_);
    ln_bf16_kernel<<<32 * CT_, 256, 0, stream>>>(text_emb, t_nc_g, t_nc_b, tp_nc, 256);
    // key = concat([nc@kc+b, nx@kx+b]) ; mask applied in column softmax
    launch_gemm(stream, tp_nc, wb_t_kc, t_kc_b, tp_key, nullptr, nullptr, 0,
                CT_, D_, 256, 256, D_, D_,
                (long long)CT_ * 256, 0, 0, 0, (long long)NT * D_, 0, 1, 32, 0, 0, 0);
    launch_gemm(stream, tp_nx, wb_t_kx, t_kx_b, tp_key + (long long)CT_ * D_, nullptr, nullptr, 0,
                T_, D_, D_, D_, D_, D_,
                (long long)T_ * D_, 0, 0, 0, (long long)NT * D_, 0, 1, 32, 0, 0, 0);
    // v = concat([nc@vc+b, (nx@vx+b)*src_mask]) -> bf16
    launch_gemm(stream, tp_nc, wb_t_vc, t_vc_b, nullptr, tp_v, nullptr, 0,
                CT_, D_, 256, 256, D_, D_,
                (long long)CT_ * 256, 0, 0, 0, (long long)NT * D_, 0, 1, 32, 0, 0, 0);
    launch_gemm(stream, tp_nx, wb_t_vx, t_vx_b, nullptr, tp_v + (long long)CT_ * D_,
                src_mask, (long long)T_,
                T_, D_, D_, D_, D_, D_,
                (long long)T_ * D_, 0, 0, 0, (long long)NT * D_, 0, 1, 32, 0, 0, 0);
    // softmaxes
    col_softmax_bf16<<<dim3(ceilDiv(D_, 256), 32), 256, 0, stream>>>(
        tp_key, tp_ks, NT, CT_, D_, src_mask, T_, 0);
    row_softmax_bf16<<<32 * T_ * H_, 64, 0, stream>>>(
        qf, tp_qs, HD_, 0, 0, 1, 0, T_, CT_, stick_mask, src_mask);
    // attn[b,h] = k_s^T @ v  (M=64,N=64,K=589)
    launch_gemm(stream, tp_ks, tp_v, nullptr, nullptr, tp_at, nullptr, 0,
                HD_, HD_, NT, D_, D_, HD_,
                (long long)NT * D_, HD_, (long long)NT * D_, HD_,
                (long long)H_ * HD_ * HD_, (long long)HD_ * HD_, H_, 32 * H_, 1, 0, 0);
    // y = q_s @ attn  (M=512,N=64,K=64)
    launch_gemm(stream, tp_qs, tp_at, nullptr, nullptr, tp_y, nullptr, 0,
                T_, HD_, HD_, D_, HD_, D_,
                (long long)T_ * D_, HD_, (long long)H_ * HD_ * HD_, (long long)HD_ * HD_,
                (long long)T_ * D_, HD_, H_, 32 * H_, 0, 0, 0);
    // query[:32] += y @ py_W + py_b
    launch_gemm(stream, tp_y, wb_t_py, t_py_b, qf, nullptr, nullptr, 0,
                32 * T_, D_, D_, D_, D_, D_, 0, 0, 0, 0, 0, 0, 1, 1, 0, 0, 1);

    // ---------------- stick branch (std attention, b=16..47) ----------------
    hipMemcpyAsync(sp_x, x + (long long)16 * T_ * D_, (size_t)32 * T_ * D_ * 4,
                   hipMemcpyDeviceToDevice, stream);
    cvt(locus_emb + (long long)16 * T_ * LD_, sp_loc, (long long)32 * T_ * LD_);
    launch_gemm(stream, sp_loc, wb_ql, p_ql_b, sp_x, nullptr, nullptr, 0,
                32 * T_, D_, LD_, LD_, D_, D_, 0, 0, 0, 0, 0, 0, 1, 1, 0, 0, 1);
    ln_bf16_kernel<<<32 * T_, 256, 0, stream>>>(sp_x, s_nx_g, s_nx_b, sp_nx, D_);
    ln_bf16_kernel<<<32 * CS_, 256, 0, stream>>>(stick_emb + (long long)16 * CS_ * 256,
                                                 s_nc_g, s_nc_b, sp_nc, 256);
    // k = concat([nc@kc+b, nx@kx+b]) -> bf16
    launch_gemm(stream, sp_nc, wb_s_kc, s_kc_b, nullptr, sp_k, nullptr, 0,
                CS_, D_, 256, 256, D_, D_,
                (long long)CS_ * 256, 0, 0, 0, (long long)NS * D_, 0, 1, 32, 0, 0, 0);
    launch_gemm(stream, sp_nx, wb_s_kx, s_kx_b, nullptr, sp_k + (long long)CS_ * D_, nullptr, 0,
                T_, D_, D_, D_, D_, D_,
                (long long)T_ * D_, 0, 0, 0, (long long)NS * D_, 0, 1, 32, 0, 0, 0);
    // v = concat([(nc@vc+b)*stick_mask, (nx@vx+b)*src_mask]) -> bf16
    launch_gemm(stream, sp_nc, wb_s_vc, s_vc_b, nullptr, sp_v,
                stick_mask + (long long)16 * CS_, (long long)CS_,
                CS_, D_, 256, 256, D_, D_,
                (long long)CS_ * 256, 0, 0, 0, (long long)NS * D_, 0, 1, 32, 0, 0, 0);
    launch_gemm(stream, sp_nx, wb_s_vx, s_vx_b, nullptr, sp_v + (long long)CS_ * D_,
                src_mask + (long long)16 * T_, (long long)T_,
                T_, D_, D_, D_, D_, D_,
                (long long)T_ * D_, 0, 0, 0, (long long)NS * D_, 0, 1, 32, 0, 0, 0);
    // chunked std attention over z = (b,h), 16 z per chunk (2 batches)
    for (int zc = 0; zc < 16; ++zc) {
        int z0 = zc * 16;
        int b0 = z0 / H_;   // local batch 0..31
        // scores = q @ k^T  (M=512, N=576, K=64)
        launch_gemm(stream, s_qb + (long long)b0 * T_ * D_, sp_k + (long long)b0 * NS * D_,
                    nullptr, sp_sc, nullptr, nullptr, 0,
                    T_, NS, HD_, D_, D_, NS,
                    (long long)T_ * D_, HD_, (long long)NS * D_, HD_,
                    (long long)H_ * T_ * NS, (long long)T_ * NS, H_, 16, 0, 1, 0);
        // softmax(+amask) -> bf16
        row_softmax_bf16<<<16 * T_, 256, 0, stream>>>(
            sp_sc, sp_a, NS, 1, CS_, H_ * T_, 16 + b0, T_, CS_, stick_mask, src_mask);
        // y = a @ v  (M=512, N=64, K=576)
        launch_gemm(stream, sp_a, sp_v + (long long)b0 * NS * D_, nullptr,
                    nullptr, sp_y + (long long)b0 * T_ * D_, nullptr, 0,
                    T_, HD_, NS, NS, D_, D_,
                    (long long)H_ * T_ * NS, (long long)T_ * NS,
                    (long long)NS * D_, HD_, (long long)T_ * D_, HD_, H_, 16, 0, 0, 0);
    }
    // query[16:48] += y @ py_W + py_b
    launch_gemm(stream, sp_y, wb_s_py, s_py_b, qf + (long long)16 * T_ * D_, nullptr, nullptr, 0,
                32 * T_, D_, D_, D_, D_, D_, 0, 0, 0, 0, 0, 0, 1, 1, 0, 0, 1);

    // ================= phase group 2: self attention + stylization ==========
    cur = phaseBase;
    ushort_t* yp_nx  = (ushort_t*)alloc((size_t)B_ * T_ * D_ * 2);
    float*    yp_key = (float*)   alloc((size_t)B_ * T_ * D_ * 4);
    ushort_t* yp_ks  = (ushort_t*)alloc((size_t)B_ * T_ * D_ * 2);
    ushort_t* yp_v   = (ushort_t*)alloc((size_t)B_ * T_ * D_ * 2);
    ushort_t* yp_qs  = (ushort_t*)alloc((size_t)B_ * T_ * D_ * 2);
    ushort_t* yp_at  = (ushort_t*)alloc((size_t)512 * HD_ * HD_ * 2);
    ushort_t* yp_y   = (ushort_t*)alloc((size_t)B_ * T_ * D_ * 2);
    float*    qy     = (float*)   alloc((size_t)B_ * T_ * D_ * 4);
    ushort_t* so     = (ushort_t*)alloc((size_t)B_ * TE_ * 2);
    float*    ef     = (float*)   alloc((size_t)B_ * 2 * D_ * 4);
    ushort_t* hmod   = (ushort_t*)alloc((size_t)B_ * T_ * D_ * 2);
    float*    otmp   = (float*)   alloc((size_t)B_ * T_ * D_ * 4);

    ln_bf16_kernel<<<B_ * T_, 256, 0, stream>>>(x, y_nx_g, y_nx_b, yp_nx, D_);
    launch_gemm(stream, yp_nx, wb_y_kx, y_kx_b, yp_key, nullptr, nullptr, 0,
                B_ * T_, D_, D_, D_, D_, D_, 0, 0, 0, 0, 0, 0, 1, 1, 0, 0, 0);
    launch_gemm(stream, yp_nx, wb_y_vx, y_vx_b, nullptr, yp_v, src_mask, 0,
                B_ * T_, D_, D_, D_, D_, D_, 0, 0, 0, 0, 0, 0, 1, 1, 0, 0, 0);
    col_softmax_bf16<<<dim3(ceilDiv(D_, 256), B_), 256, 0, stream>>>(
        yp_key, yp_ks, T_, 0, D_, src_mask, T_, 0);
    row_softmax_bf16<<<B_ * T_ * H_, 64, 0, stream>>>(
        qf, yp_qs, HD_, 0, 0, 1, 0, T_, CS_, stick_mask, src_mask);
    launch_gemm(stream, yp_ks, yp_v, nullptr, nullptr, yp_at, nullptr, 0,
                HD_, HD_, T_, D_, D_, HD_,
                (long long)T_ * D_, HD_, (long long)T_ * D_, HD_,
                (long long)H_ * HD_ * HD_, (long long)HD_ * HD_, H_, B_ * H_, 1, 0, 0);
    launch_gemm(stream, yp_qs, yp_at, nullptr, nullptr, yp_y, nullptr, 0,
                T_, HD_, HD_, D_, HD_, D_,
                (long long)T_ * D_, HD_, (long long)H_ * HD_ * HD_, (long long)HD_ * HD_,
                (long long)T_ * D_, HD_, H_, B_ * H_, 0, 0, 0);
    launch_gemm(stream, yp_y, wb_y_py, y_py_b, qy, nullptr, nullptr, 0,
                B_ * T_, D_, D_, D_, D_, D_, 0, 0, 0, 0, 0, 0, 1, 1, 0, 0, 0);

    // stylization
    silu_bf16<<<dim3((unsigned)(((long long)B_ * TE_ + 255) / 256)), 256, 0, stream>>>(
        other_emb, so, (long long)B_ * TE_);
    launch_gemm(stream, so, wb_e, p_e_b, ef, nullptr, nullptr, 0,
                B_, 2 * D_, TE_, TE_, 2 * D_, 2 * D_, 0, 0, 0, 0, 0, 0, 1, 1, 0, 0, 0);
    modulate_silu_bf16<<<B_ * T_, 256, 0, stream>>>(qy, ef, p_n_g, p_n_b, hmod, T_);
    launch_gemm(stream, hmod, wb_o, p_o_b, otmp, nullptr, nullptr, 0,
                B_ * T_, D_, D_, D_, D_, D_, 0, 0, 0, 0, 0, 0, 1, 1, 0, 0, 0);
    long long ntot = (long long)B_ * T_ * D_;
    add_kernel<<<dim3((unsigned)((ntot + 255) / 256)), 256, 0, stream>>>(
        x, otmp, (float*)d_out, ntot);
}
